// MogLSTM_69269232549936
// MI455X (gfx1250) — matrix operand order repaired and checked by reading, not compiled, not run on hardware
//
#include <hip/hip_runtime.h>

typedef __attribute__((ext_vector_type(16))) __bf16    v16bf;
typedef __attribute__((ext_vector_type(8)))  float     v8f;
typedef __attribute__((ext_vector_type(4)))  unsigned  v4u;

#define B_   256
#define S_   256
#define DIN  300
#define D2   300
#define D2P  320   // K-padded D2 (mult of 32)
#define D2N  304   // N-padded D2 (mult of 16)
#define H_   512
#define G4   2048  // 4*H

// Native bf16 converts (hardware v_cvt_* on gfx1250; round-to-nearest-even)
__device__ __forceinline__ unsigned short f2bf(float f) {
  union { __bf16 h; unsigned short u; } v; v.h = (__bf16)f; return v.u;
}
__device__ __forceinline__ float bf2f(unsigned short b) {
  union { unsigned short u; __bf16 h; } v; v.u = b; return (float)v.h;
}
__device__ __forceinline__ float sigm(float x) { return 1.0f / (1.0f + __expf(-x)); }

union ABu { v16bf v; v4u q[2]; unsigned u[8]; };

// A tile (16 rows x 32 K), row-major bf16 source (LDS or global), stride in elems.
// Lane l: M = l&15; lanes 0-15 hold K {0..7,16..23}, lanes 16-31 K {8..15,24..31}.
__device__ __forceinline__ v16bf loadA_bf(const unsigned short* p, int stride, int kb, int lane) {
  int m = lane & 15;
  int ofs = (lane >> 4) << 3;
  const unsigned short* b = p + m * stride + kb + ofs;
  ABu u;
  u.q[0] = *(const v4u*)(b);        // K ofs..ofs+7
  u.q[1] = *(const v4u*)(b + 16);   // K ofs+16..ofs+23
  return u.v;
}

// B tile (32 K x 16 N) from weight stored transposed as [N][Kpad] bf16.
// Lane l: N = nb + (l&15); lanes 0-15 hold K {0..7,16..23}, lanes 16-31 K {8..15,24..31}.
__device__ __forceinline__ v16bf loadB_tr(const unsigned short* p, int stride, int nb, int kb, int lane) {
  const unsigned short* b = p + (size_t)(nb + (lane & 15)) * stride + kb + ((lane >> 4) << 3);
  ABu u;
  u.q[0] = *(const v4u*)(b);
  u.q[1] = *(const v4u*)(b + 16);
  return u.v;
}

__device__ __forceinline__ v8f wmma_bf(v16bf a, v16bf b, v8f c) {
  return __builtin_amdgcn_wmma_f32_16x16x32_bf16(false, a, false, b, (short)0, c, false, false);
}

// ---- prep: W (K x N) fp32 -> Wt (Npad x Kpad) bf16, zero padded ----
__global__ void prep_tr(const float* __restrict__ W, unsigned short* __restrict__ out,
                        int K, int N, int Kp, int Np) {
  int total = Np * Kp;
  for (int i = blockIdx.x * blockDim.x + threadIdx.x; i < total; i += gridDim.x * blockDim.x) {
    int n = i / Kp, k = i % Kp;
    float v = (n < N && k < K) ? W[(size_t)k * N + n] : 0.f;
    out[i] = f2bf(v);
  }
}

// ---- prep: W already [N][K] fp32 -> bf16 with N padded to Np (zero rows) ----
__global__ void prep_nk(const float* __restrict__ W, unsigned short* __restrict__ out,
                        int N, int K, int Np) {
  int total = Np * K;
  for (int i = blockIdx.x * blockDim.x + threadIdx.x; i < total; i += gridDim.x * blockDim.x) {
    int n = i / K, k = i % K;
    out[i] = f2bf((n < N) ? W[(size_t)n * K + k] : 0.f);
  }
}

__global__ void prep_bsum(const float* __restrict__ a, const float* __restrict__ b,
                          float* __restrict__ o, int n) {
  int i = blockIdx.x * blockDim.x + threadIdx.x;
  if (i < n) o[i] = a[i] + b[i];
}

// ---- x (B*S, 300) fp32 -> xb (B*S, 320) bf16, zero K-pad ----
__global__ void convert_x(const float* __restrict__ x, unsigned short* __restrict__ xb) {
  int total = B_ * S_ * D2P;
  for (int i = blockIdx.x * blockDim.x + threadIdx.x; i < total; i += gridDim.x * blockDim.x) {
    int row = i / D2P, col = i - row * D2P;
    xb[i] = f2bf((col < DIN) ? x[(size_t)row * DIN + col] : 0.f);
  }
}

__global__ void init_feats(unsigned* __restrict__ f) {
  int i = blockIdx.x * blockDim.x + threadIdx.x;
  if (i < B_ * 9) f[i] = 0u;   // 0.0f bits; relu'd maxima are >= 0
}

// ---- xp = tanh(x @ W_in + b_in), stored time-major (S, B, D2P) bf16 ----
__global__ __launch_bounds__(256) void proj_kernel(const unsigned short* __restrict__ xb,
                                                   const unsigned short* __restrict__ Wt,
                                                   const float* __restrict__ b_in,
                                                   unsigned short* __restrict__ xp) {
  int lane = threadIdx.x & 31;
  int wave = threadIdx.x >> 5;
  int mt   = blockIdx.x;                       // 4096 M-tiles over B*S
  const unsigned short* arow = xb + (size_t)mt * 16 * D2P;

  for (int nt = wave; nt < 20; nt += 8) {      // 20 N-tiles cover padded 320
    int n = nt * 16 + (lane & 15);
    if (nt == 19) {                            // pure padding tile -> zeros
#pragma unroll
      for (int r = 0; r < 8; ++r) {
        int gm = mt * 16 + ((lane >> 4) << 3) + r;
        xp[(size_t)(gm & 255) * (B_ * D2P) + (size_t)(gm >> 8) * D2P + n] = 0;
      }
      continue;
    }
    v8f acc = (v8f)(0.0f);
#pragma unroll
    for (int kb = 0; kb < D2P; kb += 32)
      acc = wmma_bf(loadA_bf(arow, D2P, kb, lane), loadB_tr(Wt, D2P, nt * 16, kb, lane), acc);
    float bn = (n < D2) ? b_in[n] : 0.f;
#pragma unroll
    for (int r = 0; r < 8; ++r) {
      int gm = mt * 16 + ((lane >> 4) << 3) + r;
      float v = (n < D2) ? tanhf(acc[r] + bn) : 0.f;
      xp[(size_t)(gm & 255) * (B_ * D2P) + (size_t)(gm >> 8) * D2P + n] = f2bf(v);
    }
  }
}

// ---- persistent Mogrifier-LSTM scan: 16 WGs x 16 rows, 32 waves/WG ----
__global__ __launch_bounds__(1024) void lstm_scan(const unsigned short* __restrict__ xp,
                                                  const unsigned short* __restrict__ Qt,
                                                  const unsigned short* __restrict__ Rt,
                                                  const unsigned short* __restrict__ Wiht,
                                                  const unsigned short* __restrict__ Whht,
                                                  const float* __restrict__ bsum,
                                                  unsigned short* __restrict__ hs) {
  __shared__ alignas(16) unsigned short s_ht[16 * H_];   // state ht, bf16
  __shared__ alignas(16) unsigned short s_xt[16 * D2P];  // mogrified xt, bf16
  __shared__ alignas(16) unsigned short s_h2[16 * H_];   // mogrified ht, bf16
  __shared__ float s_b[G4];

  const int tid  = threadIdx.x;
  const int wave = tid >> 5;
  const int lane = tid & 31;
  const int rowbase = blockIdx.x << 4;
  const int mlo = (lane >> 4) << 3;            // C/D row base for this lane
  const int nl  = (wave << 4) + (lane & 15);   // owned H-column

  for (int i = tid; i < 16 * H_;  i += 1024) s_ht[i] = 0;
  for (int i = tid; i < 16 * D2P; i += 1024) s_xt[i] = 0;  // keeps K-pad 304..319 zero forever
  for (int i = tid; i < G4;       i += 1024) s_b[i] = bsum[i];

  v8f ct = (v8f)(0.0f), hst = (v8f)(0.0f);     // persistent per-wave state (16 cols x 16 rows)

  for (int t = 0; t < S_; ++t) {
    __syncthreads();                            // s_ht from previous step ready

    // ---- stage A: u = ht @ Q ; xt = 2*sigmoid(u) * xp[t] ----
    if (wave < 19) {
      int nb = wave << 4;
      int n  = nb + (lane & 15);
      v8f acc = (v8f)(0.0f);
#pragma unroll
      for (int kb = 0; kb < H_; kb += 32)
        acc = wmma_bf(loadA_bf(s_ht, H_, kb, lane), loadB_tr(Qt, H_, nb, kb, lane), acc);
      const unsigned short* xpt = xp + (size_t)t * (B_ * D2P);
#pragma unroll
      for (int r = 0; r < 8; ++r) {
        int m = mlo + r;
        float v = 0.f;
        if (n < D2) v = 2.f * sigm(acc[r]) * bf2f(xpt[(size_t)(rowbase + m) * D2P + n]);
        s_xt[m * D2P + n] = f2bf(v);
      }
    }
    __syncthreads();

    // ---- stage B: v = xt @ R ; ht' = 2*sigmoid(v) * ht ----
    {
      int nb = wave << 4;
      v8f acc = (v8f)(0.0f);
#pragma unroll
      for (int kb = 0; kb < D2P; kb += 32)
        acc = wmma_bf(loadA_bf(s_xt, D2P, kb, lane), loadB_tr(Rt, D2P, nb, kb, lane), acc);
#pragma unroll
      for (int r = 0; r < 8; ++r)
        s_h2[(mlo + r) * H_ + nl] = f2bf(2.f * sigm(acc[r]) * hst[r]);
    }
    __syncthreads();

    // ---- stage C: gates = xt @ Wih + ht' @ Whh + b ; LSTM pointwise ----
    auto gate = [&](int gi) -> v8f {
      int nb = (gi << 9) + (wave << 4);
      v8f acc = (v8f)(0.0f);
#pragma unroll
      for (int kb = 0; kb < D2P; kb += 32)
        acc = wmma_bf(loadA_bf(s_xt, D2P, kb, lane), loadB_tr(Wiht, D2P, nb, kb, lane), acc);
#pragma unroll
      for (int kb = 0; kb < H_; kb += 32)
        acc = wmma_bf(loadA_bf(s_h2, H_, kb, lane), loadB_tr(Whht, H_, nb, kb, lane), acc);
      return acc;
    };
    v8f g0 = gate(0), g1 = gate(1), g2 = gate(2), g3 = gate(3);

#pragma unroll
    for (int r = 0; r < 8; ++r) {
      float ig = sigm(g0[r] + s_b[nl]);
      float fg = sigm(g1[r] + s_b[512 + nl]);
      float gg = tanhf(g2[r] + s_b[1024 + nl]);
      float og = sigm(g3[r] + s_b[1536 + nl]);
      float c  = fg * ct[r] + ig * gg;
      ct[r] = c;
      float h = og * tanhf(c);
      hst[r] = h;
      unsigned short hb = f2bf(h);
      int m = mlo + r;
      s_ht[m * H_ + nl] = hb;
      hs[((size_t)(rowbase + m) * S_ + t) * H_ + nl] = hb;
    }
  }
}

// ---- conv as WMMA GEMM: A rows are contiguous fsz*512 spans of hs (overlapping),
//      B is conv weights [16(N-pad)][fsz*512] bf16; relu+time-max via atomicMax on bits ----
__global__ __launch_bounds__(32) void conv_wmma(const unsigned short* __restrict__ hs,
                                                const unsigned short* __restrict__ cwt,
                                                const float* __restrict__ b3,
                                                const float* __restrict__ b4,
                                                const float* __restrict__ b5,
                                                unsigned* __restrict__ feats) {
  const int lane = threadIdx.x & 31;
  const int mt = blockIdx.x;                   // 4096 tiles over 65536 (b,t0) rows
  const int si = blockIdx.y;                   // 0..2 -> fsz 3,4,5
  const int fs = 3 + si;
  const int K  = fs * 512;
  const unsigned short* Bw = cwt + ((si > 0 ? 16 * 1536 : 0) + (si > 1 ? 16 * 2048 : 0));
  const float* bp = (si == 0) ? b3 : (si == 1) ? b4 : b5;

  const unsigned short* arow = hs + (size_t)mt * 16 * H_;   // A stride 512, row len K (overlap)
  v8f acc = (v8f)(0.0f);
  for (int kb = 0; kb < K; kb += 32)
    acc = wmma_bf(loadA_bf(arow, H_, kb, lane), loadB_tr(Bw, K, 0, kb, lane), acc);

  int n = lane & 15;                            // filter index (only 0..2 valid)
  if (n < 3) {
    float bias = bp[n];
    int mlo = (lane >> 4) << 3;
#pragma unroll
    for (int r = 0; r < 8; ++r) {
      int grow = mt * 16 + mlo + r;             // (b, t0) flat row
      int b  = grow >> 8;
      int t0 = grow & 255;
      if (t0 + fs <= S_) {
        float v = fmaxf(acc[r] + bias, 0.f);    // relu -> nonneg, bits order-preserving
        atomicMax(&feats[b * 9 + si * 3 + n], __float_as_uint(v));
      }
    }
  }
}

// ---- final: out = feats @ lin_w + lin_b ----
__global__ __launch_bounds__(256) void final_head(const unsigned* __restrict__ feats,
                                                  const float* __restrict__ lw,
                                                  const float* __restrict__ lb,
                                                  float* __restrict__ out) {
  int b = threadIdx.x;
  float a0 = lb[0], a1 = lb[1];
#pragma unroll
  for (int i = 0; i < 9; ++i) {
    float f = __uint_as_float(feats[b * 9 + i]);
    a0 += f * lw[i * 2 + 0];
    a1 += f * lw[i * 2 + 1];
  }
  out[b * 2 + 0] = a0;
  out[b * 2 + 1] = a1;
}

extern "C" void kernel_launch(void* const* d_in, const int* in_sizes, int n_in,
                              void* d_out, int out_size, void* d_ws, size_t ws_size,
                              hipStream_t stream) {
  (void)in_sizes; (void)n_in; (void)out_size; (void)ws_size;
  const float* x     = (const float*)d_in[0];
  const float* W_in  = (const float*)d_in[1];
  const float* b_in  = (const float*)d_in[2];
  const float* Wih   = (const float*)d_in[3];
  const float* Whh   = (const float*)d_in[4];
  const float* bih   = (const float*)d_in[5];
  const float* bhh   = (const float*)d_in[6];
  const float* Q     = (const float*)d_in[7];
  const float* R     = (const float*)d_in[8];
  const float* lin_w = (const float*)d_in[9];
  const float* lin_b = (const float*)d_in[10];
  const float* w3    = (const float*)d_in[11];
  const float* b3    = (const float*)d_in[12];
  const float* w4    = (const float*)d_in[13];
  const float* b4    = (const float*)d_in[14];
  const float* w5    = (const float*)d_in[15];
  const float* b5    = (const float*)d_in[16];
  float* out = (float*)d_out;

  char* ws = (char*)d_ws;
  size_t off = 0;
  auto alloc = [&](size_t bytes) { char* p = ws + off; off += (bytes + 255) & ~(size_t)255; return p; };

  unsigned short* xb   = (unsigned short*)alloc((size_t)B_ * S_ * D2P * 2);  // x in bf16, K-padded
  unsigned short* xp   = (unsigned short*)alloc((size_t)S_ * B_ * D2P * 2);  // projected, time-major
  unsigned short* hsb  = (unsigned short*)alloc((size_t)B_ * S_ * H_  * 2);  // hidden seq (b,t,h)
  unsigned short* Wint = (unsigned short*)alloc((size_t)D2N * D2P * 2);
  unsigned short* Qt   = (unsigned short*)alloc((size_t)D2N * H_  * 2);
  unsigned short* Rt   = (unsigned short*)alloc((size_t)H_  * D2P * 2);
  unsigned short* Wiht = (unsigned short*)alloc((size_t)G4  * D2P * 2);
  unsigned short* Whht = (unsigned short*)alloc((size_t)G4  * H_  * 2);
  unsigned short* cwt  = (unsigned short*)alloc((size_t)16 * (1536 + 2048 + 2560) * 2);
  float*          bsum = (float*)alloc((size_t)G4 * 4);
  unsigned*       feats = (unsigned*)alloc((size_t)B_ * 9 * 4);

  prep_tr<<<256,  256, 0, stream>>>(W_in, Wint, DIN, D2, D2P, D2N);
  prep_tr<<<512,  256, 0, stream>>>(Q,    Qt,   H_,  D2, H_,  D2N);
  prep_tr<<<512,  256, 0, stream>>>(R,    Rt,   D2,  H_, D2P, H_);
  prep_tr<<<1024, 256, 0, stream>>>(Wih,  Wiht, D2,  G4, D2P, G4);
  prep_tr<<<1024, 256, 0, stream>>>(Whh,  Whht, H_,  G4, H_,  G4);
  prep_nk<<<96,   256, 0, stream>>>(w3, cwt,                       3, 1536, 16);
  prep_nk<<<128,  256, 0, stream>>>(w4, cwt + 16 * 1536,           3, 2048, 16);
  prep_nk<<<160,  256, 0, stream>>>(w5, cwt + 16 * (1536 + 2048),  3, 2560, 16);
  prep_bsum<<<8,  256, 0, stream>>>(bih, bhh, bsum, G4);
  convert_x<<<8192, 256, 0, stream>>>(x, xb);
  init_feats<<<9, 256, 0, stream>>>(feats);

  proj_kernel<<<(B_ * S_) / 16, 256, 0, stream>>>(xb, Wint, b_in, xp);
  lstm_scan<<<B_ / 16, 1024, 0, stream>>>(xp, Qt, Rt, Wiht, Whht, bsum, hsb);
  conv_wmma<<<dim3((B_ * S_) / 16, 3), 32, 0, stream>>>(hsb, cwt, b3, b4, b5, feats);
  final_head<<<1, 256, 0, stream>>>(feats, lin_w, lin_b, out);
}
